// AttentionHead_19499151524546
// MI455X (gfx1250) — compile-verified, matrix operand order checked by reading, not run on hardware
//
#include <hip/hip_runtime.h>

#define B_ 4
#define S_ 2048
#define D_ 1024

// Projection GEMM tiling (fused QKV, 3 accumulator sets -> modest wave tile)
#define PM 128
#define PN 64
#define PK 32

// Attention GEMM tiling (scores, PV): 64x64 wave tile, 1:1 ds:wmma ratio
#define GM 256
#define GN 128
#define GK 64

typedef __bf16 bf16;
typedef __attribute__((ext_vector_type(16))) __bf16 bf16x16;
typedef __attribute__((ext_vector_type(8)))  __bf16 bf16x8;
typedef __attribute__((ext_vector_type(8)))  float  f32x8;
typedef __attribute__((ext_vector_type(4)))  float  f32x4;

__device__ __forceinline__ f32x8 wmma_bf16(bf16x16 a, bf16x16 b, f32x8 c) {
  // D = A(16x32 bf16) * B(32x16 bf16) + C(16x16 f32)
  return __builtin_amdgcn_wmma_f32_16x16x32_bf16(false, a, false, b, (short)0, c,
                                                 false, false);
}

// Load one 16x32 bf16 fragment from an LDS tile stored row-major, leading dim ld.
// CDNA5 16-bit A layout (wave32):
//   lanes 0-15 : row = lane,    K = 0..7 and 16..23
//   lanes 16-31: row = lane-16, K = 8..15 and 24..31
// Same layout serves the B operand when the tile holds B^T (rows = N, cols = K).
__device__ __forceinline__ bf16x16 ld_frag(const bf16* base, int ld) {
  const int lane = threadIdx.x & 31;
  const bf16* p = base + (lane & 15) * ld + ((lane >> 4) << 3);
  bf16x8 lo = *(const bf16x8*)(p);
  bf16x8 hi = *(const bf16x8*)(p + 16);
  return __builtin_shufflevector(lo, hi, 0, 1, 2, 3, 4, 5, 6, 7,
                                 8, 9, 10, 11, 12, 13, 14, 15);
}

// ---------------------------------------------------------------------------
// 1) Weight transpose + fp32 -> bf16:  Wt[n][k] = (bf16) W[k][n]
// ---------------------------------------------------------------------------
__global__ __launch_bounds__(256) void k_wtrans(const float* __restrict__ W,
                                                bf16* __restrict__ Wt) {
  __shared__ float t[32][33];
  const int tx = threadIdx.x & 31;
  const int ty = threadIdx.x >> 5;          // 0..7
  const int bx = blockIdx.x * 32;           // n tile
  const int by = blockIdx.y * 32;           // k tile
#pragma unroll
  for (int i = 0; i < 4; ++i)
    t[ty + i * 8][tx] = W[(size_t)(by + ty + i * 8) * D_ + (bx + tx)];
  __syncthreads();
#pragma unroll
  for (int i = 0; i < 4; ++i)
    Wt[(size_t)(bx + ty + i * 8) * D_ + (by + tx)] = (bf16)t[tx][ty + i * 8];
}

// ---------------------------------------------------------------------------
// 2) Fused QKV projection: one 128x64 tile of Q, K, V per block, sharing the
//    embedding tile.  Q,K stored bf16 row-major [b*S+s][d]; V stored bf16
//    transposed per batch: Vt[b][d][s]  (so PV GEMM B-operand is contiguous).
// ---------------------------------------------------------------------------
__global__ __launch_bounds__(256) void k_qkv(const float* __restrict__ E,
                                             const bf16* __restrict__ WtQ,
                                             const bf16* __restrict__ WtK,
                                             const bf16* __restrict__ WtV,
                                             bf16* __restrict__ Q,
                                             bf16* __restrict__ K,
                                             bf16* __restrict__ Vt) {
  __shared__ __align__(16) bf16 Ash[PM * PK];       // 8 KB
  __shared__ __align__(16) bf16 Bsh[3][PN * PK];    // 12 KB
  const int tid  = threadIdx.x;
  const int wave = tid >> 5;
  const int wm   = wave & 3;          // 4 wave-rows  -> 128
  const int wn   = wave >> 2;         // 2 wave-cols  -> 64
  const int m0   = blockIdx.x * PM;   // global row (b*S + s)
  const int n0   = blockIdx.y * PN;   // output feature
  const bf16* Wts[3] = {WtQ, WtK, WtV};

  f32x8 acc[3][2][2] = {};

  for (int kb = 0; kb < D_ / PK; ++kb) {
    // stage A: 128x32 fp32 -> bf16 in LDS
#pragma unroll
    for (int i = 0; i < 4; ++i) {
      int idx = (tid + i * 256) << 2;       // element idx, step 4
      int r = idx >> 5, c = idx & 31;
      f32x4 v = *(const f32x4*)(E + (size_t)(m0 + r) * D_ + kb * PK + c);
#pragma unroll
      for (int j = 0; j < 4; ++j) Ash[r * PK + c + j] = (bf16)v[j];
    }
    // stage B: three 64x32 bf16 weight tiles (already transposed)
    {
      int r = tid >> 2;
      int c = (tid & 3) << 3;
#pragma unroll
      for (int w = 0; w < 3; ++w)
        *(bf16x8*)&Bsh[w][r * PK + c] =
            *(const bf16x8*)(Wts[w] + (size_t)(n0 + r) * D_ + kb * PK + c);
    }
    __syncthreads();

    bf16x16 a0 = ld_frag(&Ash[(wm * 32 + 0) * PK], PK);
    bf16x16 a1 = ld_frag(&Ash[(wm * 32 + 16) * PK], PK);
#pragma unroll
    for (int w = 0; w < 3; ++w) {
      bf16x16 b0 = ld_frag(&Bsh[w][(wn * 32 + 0) * PK], PK);
      bf16x16 b1 = ld_frag(&Bsh[w][(wn * 32 + 16) * PK], PK);
      acc[w][0][0] = wmma_bf16(a0, b0, acc[w][0][0]);
      acc[w][0][1] = wmma_bf16(a0, b1, acc[w][0][1]);
      acc[w][1][0] = wmma_bf16(a1, b0, acc[w][1][0]);
      acc[w][1][1] = wmma_bf16(a1, b1, acc[w][1][1]);
    }
    __syncthreads();
  }

  const int lane = tid & 31;
  const int nn = lane & 15;
  const int mb = (lane >> 4) << 3;      // C/D layout: lanes 16-31 hold M=8..15
  const int b  = m0 / S_;
  const int s0 = m0 % S_;
#pragma unroll
  for (int ti = 0; ti < 2; ++ti)
#pragma unroll
    for (int tj = 0; tj < 2; ++tj) {
      const int mrow = wm * 32 + ti * 16 + mb;
      const int ncol = wn * 32 + tj * 16 + nn;
#pragma unroll
      for (int i = 0; i < 8; ++i) {
        size_t off = (size_t)(m0 + mrow + i) * D_ + (n0 + ncol);
        Q[off] = (bf16)acc[0][ti][tj][i];
        K[off] = (bf16)acc[1][ti][tj][i];
      }
      bf16x8 pk;
#pragma unroll
      for (int i = 0; i < 8; ++i) pk[i] = (bf16)acc[2][ti][tj][i];
      // V transposed: rows of the accumulator are contiguous along s
      *(bf16x8*)(Vt + (size_t)b * D_ * S_ + (size_t)(n0 + ncol) * S_ +
                 (s0 + mrow)) = pk;
    }
}

// ---------------------------------------------------------------------------
// 3) scores = (Q K^T) / 32 with causal mask, fp32 into workspace (fits in L2).
//    256x128 block tile, 64x64 wave tile (4x4 WMMA frags), BK=64.
// ---------------------------------------------------------------------------
__global__ __launch_bounds__(256) void k_scores(const bf16* __restrict__ Qg,
                                                const bf16* __restrict__ Kg,
                                                float* __restrict__ SC) {
  const int tid = threadIdx.x;
  const int b   = blockIdx.z;
  const int m0  = blockIdx.x * GM;     // query rows
  const int n0  = blockIdx.y * GN;     // key cols
  float* sc = SC + (size_t)b * S_ * S_;
  const float NEG_INF = -__builtin_inff();

  if (n0 >= m0 + GM) {                 // tile entirely above the diagonal
    f32x4 fill = {NEG_INF, NEG_INF, NEG_INF, NEG_INF};
#pragma unroll
    for (int i = 0; i < 32; ++i) {
      int idx = tid + i * 256;         // vec4 chunk id, 8192 total
      int r = idx >> 5, c = (idx & 31) << 2;
      *(f32x4*)&sc[(size_t)(m0 + r) * S_ + (n0 + c)] = fill;
    }
    return;
  }

  __shared__ __align__(16) bf16 Ash[GM * GK];   // 32 KB
  __shared__ __align__(16) bf16 Bsh[GN * GK];   // 16 KB
  const bf16* A  = Qg + (size_t)b * S_ * D_;
  const bf16* Bt = Kg + (size_t)b * S_ * D_;    // K row-major == B^T layout
  const int wave = tid >> 5;
  const int wm = wave >> 1;            // 4 wave-rows of 64
  const int wn = wave & 1;             // 2 wave-cols of 64
  f32x8 acc[4][4] = {};

  for (int kb = 0; kb < D_ / GK; ++kb) {
    if (kb + 1 < D_ / GK) {            // prefetch next k-tile rows into L2/L0
      __builtin_prefetch(A + (size_t)(m0 + tid) * D_ + (kb + 1) * GK, 0, 0);
      if (tid < GN)
        __builtin_prefetch(Bt + (size_t)(n0 + tid) * D_ + (kb + 1) * GK, 0, 0);
    }
#pragma unroll
    for (int j = 0; j < 8; ++j) {      // A: 2048 chunks of 8 bf16
      int ch = tid + j * 256;
      int r = ch >> 3, c = (ch & 7) << 3;
      *(bf16x8*)&Ash[r * GK + c] =
          *(const bf16x8*)(A + (size_t)(m0 + r) * D_ + kb * GK + c);
    }
#pragma unroll
    for (int j = 0; j < 4; ++j) {      // B: 1024 chunks
      int ch = tid + j * 256;
      int r = ch >> 3, c = (ch & 7) << 3;
      *(bf16x8*)&Bsh[r * GK + c] =
          *(const bf16x8*)(Bt + (size_t)(n0 + r) * D_ + kb * GK + c);
    }
    __syncthreads();
#pragma unroll
    for (int kh = 0; kh < 2; ++kh) {
      bf16x16 bfr[4];
#pragma unroll
      for (int ni = 0; ni < 4; ++ni)
        bfr[ni] = ld_frag(&Bsh[(wn * 64 + ni * 16) * GK + kh * 32], GK);
#pragma unroll
      for (int mi = 0; mi < 4; ++mi) {
        bf16x16 a = ld_frag(&Ash[(wm * 64 + mi * 16) * GK + kh * 32], GK);
#pragma unroll
        for (int ni = 0; ni < 4; ++ni)
          acc[mi][ni] = wmma_bf16(a, bfr[ni], acc[mi][ni]);
      }
    }
    __syncthreads();
  }

  const int lane = tid & 31;
  const int nn = lane & 15, mb = (lane >> 4) << 3;
  const float scale = 0.03125f;        // 1/sqrt(1024)
#pragma unroll
  for (int mi = 0; mi < 4; ++mi)
#pragma unroll
    for (int ni = 0; ni < 4; ++ni)
#pragma unroll
      for (int i = 0; i < 8; ++i) {
        int gm = m0 + wm * 64 + mi * 16 + mb + i;
        int gn = n0 + wn * 64 + ni * 16 + nn;
        sc[(size_t)gm * S_ + gn] =
            (gn <= gm) ? acc[mi][ni][i] * scale : NEG_INF;
      }
}

// ---------------------------------------------------------------------------
// 4) Row softmax, writing bf16 probs in-place over the fp32 score row.
//    One 256-thread block per row (B*S rows), 8 values per thread.
// ---------------------------------------------------------------------------
__global__ __launch_bounds__(256) void k_softmax(float* __restrict__ SC) {
  const int tid = threadIdx.x;
  const size_t row = blockIdx.x;
  float* src = SC + row * (size_t)S_;
  bf16* dst = (bf16*)src;              // first half of the fp32 row
  __shared__ float red[256];
  float v[8];
  float mx = -__builtin_inff();
#pragma unroll
  for (int i = 0; i < 8; ++i) {
    v[i] = src[tid + i * 256];
    mx = fmaxf(mx, v[i]);
  }
  red[tid] = mx;
  __syncthreads();
  for (int s = 128; s > 0; s >>= 1) {
    if (tid < s) red[tid] = fmaxf(red[tid], red[tid + s]);
    __syncthreads();
  }
  mx = red[0];
  __syncthreads();
  float sum = 0.f;
#pragma unroll
  for (int i = 0; i < 8; ++i) {
    v[i] = __expf(v[i] - mx);          // exp(-inf - mx) == 0 for masked
    sum += v[i];
  }
  red[tid] = sum;
  __syncthreads();
  for (int s = 128; s > 0; s >>= 1) {
    if (tid < s) red[tid] += red[tid + s];
    __syncthreads();
  }
  const float inv = 1.f / red[0];
#pragma unroll
  for (int i = 0; i < 8; ++i) dst[tid + i * 256] = (bf16)(v[i] * inv);
}

// ---------------------------------------------------------------------------
// 5) O = P @ V.  P is bf16 with row pitch 2*S (in-place over fp32 rows);
//    V^T is contiguous in k.  Causal: k-loop capped at the query block.
//    Same 256x128 / 64x64-wave tiling as k_scores.
// ---------------------------------------------------------------------------
__global__ __launch_bounds__(256) void k_pv(const float* __restrict__ SC,
                                            const bf16* __restrict__ Vt,
                                            float* __restrict__ O) {
  const int tid = threadIdx.x;
  const int b   = blockIdx.z;
  const int m0  = blockIdx.x * GM;     // query rows
  const int n0  = blockIdx.y * GN;     // output features
  const bf16* P  = (const bf16*)(SC + (size_t)b * S_ * S_);
  const int ldp  = 2 * S_;             // probs row pitch in bf16 elements
  const bf16* Bt = Vt + (size_t)b * D_ * S_;
  __shared__ __align__(16) bf16 Ash[GM * GK];   // 32 KB
  __shared__ __align__(16) bf16 Bsh[GN * GK];   // 16 KB
  const int wave = tid >> 5;
  const int wm = wave >> 1;
  const int wn = wave & 1;
  f32x8 acc[4][4] = {};
  const int kTiles = (m0 + GM) / GK;   // keys <= m0+GM-1 (rest are zeros)

  for (int kb = 0; kb < kTiles; ++kb) {
    if (kb + 1 < kTiles) {
      __builtin_prefetch(P + (size_t)(m0 + tid) * ldp + (kb + 1) * GK, 0, 0);
      if (tid < GN)
        __builtin_prefetch(Bt + (size_t)(n0 + tid) * S_ + (kb + 1) * GK, 0, 0);
    }
#pragma unroll
    for (int j = 0; j < 8; ++j) {
      int ch = tid + j * 256;
      int r = ch >> 3, c = (ch & 7) << 3;
      *(bf16x8*)&Ash[r * GK + c] =
          *(const bf16x8*)(P + (size_t)(m0 + r) * ldp + kb * GK + c);
    }
#pragma unroll
    for (int j = 0; j < 4; ++j) {
      int ch = tid + j * 256;
      int r = ch >> 3, c = (ch & 7) << 3;
      *(bf16x8*)&Bsh[r * GK + c] =
          *(const bf16x8*)(Bt + (size_t)(n0 + r) * S_ + kb * GK + c);
    }
    __syncthreads();
#pragma unroll
    for (int kh = 0; kh < 2; ++kh) {
      bf16x16 bfr[4];
#pragma unroll
      for (int ni = 0; ni < 4; ++ni)
        bfr[ni] = ld_frag(&Bsh[(wn * 64 + ni * 16) * GK + kh * 32], GK);
#pragma unroll
      for (int mi = 0; mi < 4; ++mi) {
        bf16x16 a = ld_frag(&Ash[(wm * 64 + mi * 16) * GK + kh * 32], GK);
#pragma unroll
        for (int ni = 0; ni < 4; ++ni)
          acc[mi][ni] = wmma_bf16(a, bfr[ni], acc[mi][ni]);
      }
    }
    __syncthreads();
  }

  const int lane = tid & 31;
  const int nn = lane & 15, mb = (lane >> 4) << 3;
#pragma unroll
  for (int mi = 0; mi < 4; ++mi)
#pragma unroll
    for (int ni = 0; ni < 4; ++ni)
#pragma unroll
      for (int i = 0; i < 8; ++i) {
        int gm = m0 + wm * 64 + mi * 16 + mb + i;
        int gn = n0 + wn * 64 + ni * 16 + nn;
        O[((size_t)b * S_ + gm) * D_ + gn] = acc[mi][ni][i];
      }
}

// ---------------------------------------------------------------------------
extern "C" void kernel_launch(void* const* d_in, const int* in_sizes, int n_in,
                              void* d_out, int out_size, void* d_ws,
                              size_t ws_size, hipStream_t stream) {
  (void)in_sizes; (void)n_in; (void)out_size; (void)ws_size;
  const float* E  = (const float*)d_in[0];
  const float* WQ = (const float*)d_in[1];
  const float* WK = (const float*)d_in[2];
  const float* WV = (const float*)d_in[3];
  float* out = (float*)d_out;

  // Workspace layout (bytes):
  //   Wt(Q,K,V)  bf16 : 3 * D*D*2      =  6 MB
  //   Q, K       bf16 : 2 * B*S*D*2    = 32 MB
  //   V^T        bf16 : 1 * B*S*D*2    = 16 MB
  //   scores     fp32 : B*S*S*4        = 67 MB  (probs bf16 written in-place)
  char* ws = (char*)d_ws;
  bf16* WtQ = (bf16*)ws;
  bf16* WtK = WtQ + (size_t)D_ * D_;
  bf16* WtV = WtK + (size_t)D_ * D_;
  bf16* Qb  = WtV + (size_t)D_ * D_;
  bf16* Kb  = Qb + (size_t)B_ * S_ * D_;
  bf16* Vt  = Kb + (size_t)B_ * S_ * D_;
  float* SC = (float*)(Vt + (size_t)B_ * S_ * D_);

  dim3 t(256);
  k_wtrans<<<dim3(D_ / 32, D_ / 32), t, 0, stream>>>(WQ, WtQ);
  k_wtrans<<<dim3(D_ / 32, D_ / 32), t, 0, stream>>>(WK, WtK);
  k_wtrans<<<dim3(D_ / 32, D_ / 32), t, 0, stream>>>(WV, WtV);
  k_qkv<<<dim3(B_ * S_ / PM, D_ / PN), t, 0, stream>>>(E, WtQ, WtK, WtV, Qb, Kb,
                                                       Vt);
  k_scores<<<dim3(S_ / GM, S_ / GN, B_), t, 0, stream>>>(Qb, Kb, SC);
  k_softmax<<<dim3(B_ * S_), t, 0, stream>>>(SC);
  k_pv<<<dim3(S_ / GM, D_ / GN, B_), t, 0, stream>>>(SC, Vt, out);
}
